// GNNFusion_30726196035938
// MI455X (gfx1250) — matrix operand order repaired and checked by reading, not compiled, not run on hardware
//
#include <hip/hip_runtime.h>
#include <hip/hip_bf16.h>

typedef __bf16 bf16_t;
typedef __attribute__((ext_vector_type(16))) __bf16 v16bf;
typedef __attribute__((ext_vector_type(8)))  __bf16 v8bf;
typedef __attribute__((ext_vector_type(8)))  float  v8f;

#define H 256
#define GG 512
#define GF 128
#define SE 768
#define NEG_SLOPE 0.01f
#define EPS 1e-5f

__device__ __forceinline__ float leakyf(float v) { return v > 0.f ? v : NEG_SLOPE * v; }
__device__ __forceinline__ float sigmf(float v) { return 1.f / (1.f + __expf(-v)); }

// ---------------- elementwise conversions ----------------

__global__ void k_cvt(const float* __restrict__ in, bf16_t* __restrict__ out, int n) {
  int t = blockIdx.x * blockDim.x + threadIdx.x;
  if (t < n) out[t] = (bf16_t)in[t];
}

__global__ void k_leaky_cvt(const float* __restrict__ agg, float* __restrict__ hf,
                            bf16_t* __restrict__ hb, int n) {
  int t = blockIdx.x * blockDim.x + threadIdx.x;
  if (t < n) {
    float v = leakyf(agg[t]);
    hf[t] = v;
    hb[t] = (bf16_t)v;
  }
}

// ---------------- degree / normalization ----------------

__global__ void k_fill1(float* __restrict__ p, int n) {
  int t = blockIdx.x * blockDim.x + threadIdx.x;
  if (t < n) p[t] = 1.0f;  // self-loop contributes 1 to every degree
}

__global__ void k_deg_edges(const int* __restrict__ col, float* __restrict__ deg, int e) {
  int t = blockIdx.x * blockDim.x + threadIdx.x;
  if (t < e) unsafeAtomicAdd(&deg[col[t]], 1.0f);
}

__global__ void k_dis(float* __restrict__ dis, int n) {
  int t = blockIdx.x * blockDim.x + threadIdx.x;
  if (t < n) dis[t] = rsqrtf(dis[t]);  // deg >= 1 always (self loops)
}

// ---------------- weight repack for WMMA B fragments ----------------
// Dense 16-bit B 32x16 layout: lane half h, elem e<8 -> K=8h+e, elem e>=8 -> K=16+8h+(e-8).
// Wp layout: [kChunk(8)][nTile(16)][lane(32)][elem(16)] bf16, contiguous 32B per lane.
__global__ void k_pack_w(const float* __restrict__ W, bf16_t* __restrict__ Wp) {
  int t = blockIdx.x * blockDim.x + threadIdx.x;  // 8*16*32*16 = 65536
  if (t >= 8 * 16 * 32 * 16) return;
  int e = t & 15;
  int L = (t >> 4) & 31;
  int nt = (t >> 9) & 15;
  int c = t >> 13;
  int n = nt * 16 + (L & 15);
  int k = c * 32 + ((L >> 4) * 8) + (e & 7) + ((e >> 3) * 16);
  Wp[t] = (bf16_t)W[k * H + n];
}

// ---------------- WMMA GEMM: C[N x 256] = A[N x 256] * W[256 x 256] ----------------
// One wave per 16x64 output strip (4 column tiles): A fragment loaded once per K-chunk
// feeds 4 independent-accumulator WMMAs (no back-to-back RAW hazard, 4x A reuse).
__global__ void k_gemm_bf16(const bf16_t* __restrict__ A, const bf16_t* __restrict__ Wp,
                            float* __restrict__ C, int mtiles) {
  int wave = (blockIdx.x * blockDim.x + threadIdx.x) >> 5;  // uniform per wave
  int lane = threadIdx.x & 31;
  int rowTile = wave >> 2;  // 4 column-groups per row tile
  int cg = wave & 3;        // column tiles cg*4 .. cg*4+3
  if (rowTile >= mtiles) return;  // wave-uniform: EXEC stays all-ones

  int m = lane & 15;
  int khalf = (lane >> 4) * 8;
  const bf16_t* arow = A + (size_t)(rowTile * 16 + m) * H + khalf;

  v8f acc0 = {}, acc1 = {}, acc2 = {}, acc3 = {};
#pragma unroll
  for (int c = 0; c < 8; ++c) {
    // A fragment (16-bit A 16x32): elems 0-7 = K khalf..khalf+7, elems 8-15 = K 16+khalf..
    const bf16_t* ap = arow + c * 32;
    v8bf a0 = *(const v8bf*)(ap);
    v8bf a1 = *(const v8bf*)(ap + 16);
    v16bf a;
#pragma unroll
    for (int i = 0; i < 8; ++i) { a[i] = a0[i]; a[i + 8] = a1[i]; }
    const bf16_t* bp = Wp + ((((size_t)c * 16 + cg * 4) * 32 + lane) << 4);
    v16bf b0 = *(const v16bf*)(bp);
    v16bf b1 = *(const v16bf*)(bp + 32 * 16);
    v16bf b2 = *(const v16bf*)(bp + 2 * 32 * 16);
    v16bf b3 = *(const v16bf*)(bp + 3 * 32 * 16);
    acc0 = __builtin_amdgcn_wmma_f32_16x16x32_bf16(false, a, false, b0, (short)0, acc0,
                                                   false, false);
    acc1 = __builtin_amdgcn_wmma_f32_16x16x32_bf16(false, a, false, b1, (short)0, acc1,
                                                   false, false);
    acc2 = __builtin_amdgcn_wmma_f32_16x16x32_bf16(false, a, false, b2, (short)0, acc2,
                                                   false, false);
    acc3 = __builtin_amdgcn_wmma_f32_16x16x32_bf16(false, a, false, b3, (short)0, acc3,
                                                   false, false);
  }
  // C/D layout: VGPR r, lane L -> M = r + 8*(L>>4), N = L&15
  int row0 = rowTile * 16 + 8 * (lane >> 4);
  int colb = cg * 64 + (lane & 15);
#pragma unroll
  for (int r = 0; r < 8; ++r) {
    float* cp = C + (size_t)(row0 + r) * H + colb;
    cp[0] = acc0[r];
    cp[16] = acc1[r];
    cp[32] = acc2[r];
    cp[48] = acc3[r];
  }
}

// ---------------- GCN aggregation ----------------

// agg[i][j] = b[j] + dis[i]^2 * h[i][j]   (bias + self-loop term)
__global__ void k_agg_init(const float* __restrict__ hf, const float* __restrict__ dis,
                           const float* __restrict__ bias, float* __restrict__ agg, int n) {
  int t = blockIdx.x * blockDim.x + threadIdx.x;
  if (t < n) {
    int i = t >> 8;
    float d = dis[i];
    agg[t] = bias[t & 255] + d * d * hf[t];
  }
}

// per real edge: agg[col] += dis[row]*dis[col] * h[row]; 32 threads/edge, 8 feats/thread
__global__ void k_agg_edges(const int* __restrict__ row, const int* __restrict__ col,
                            const float* __restrict__ dis, const float* __restrict__ hf,
                            float* __restrict__ agg, int e) {
  int t = blockIdx.x * blockDim.x + threadIdx.x;
  int ed = t >> 5;
  if (ed >= e) return;
  int j0 = (t & 31) * 8;
  int r = row[ed], c = col[ed];
  float w = dis[r] * dis[c];
  const float4* src = (const float4*)(hf + (size_t)r * H + j0);
  float4 v0 = src[0], v1 = src[1];
  float* dst = agg + (size_t)c * H + j0;
  unsafeAtomicAdd(dst + 0, w * v0.x);
  unsafeAtomicAdd(dst + 1, w * v0.y);
  unsafeAtomicAdd(dst + 2, w * v0.z);
  unsafeAtomicAdd(dst + 3, w * v0.w);
  unsafeAtomicAdd(dst + 4, w * v1.x);
  unsafeAtomicAdd(dst + 5, w * v1.y);
  unsafeAtomicAdd(dst + 6, w * v1.z);
  unsafeAtomicAdd(dst + 7, w * v1.w);
}

// ---------------- pooling ----------------

__global__ void k_pool_init(float* __restrict__ ps, float* __restrict__ cnt, int n) {
  int t = blockIdx.x * blockDim.x + threadIdx.x;
  if (t < n) ps[t] = 0.f;
  if (t < GG) cnt[t] = 0.f;
}

__global__ void k_pool(const float* __restrict__ hf, const int* __restrict__ batch,
                       float* __restrict__ ps, float* __restrict__ cnt, int n) {
  int t = blockIdx.x * blockDim.x + threadIdx.x;
  if (t >= n) return;
  int i = t >> 8, j = t & 255;
  int g = batch[i];
  unsafeAtomicAdd(&ps[(size_t)g * H + j], hf[t]);
  if (j == 0) unsafeAtomicAdd(&cnt[g], 1.0f);
}

__global__ void k_pool_fin(float* __restrict__ ps, const float* __restrict__ cnt) {
  int t = blockIdx.x * blockDim.x + threadIdx.x;  // GG*256
  ps[t] /= fmaxf(cnt[t >> 8], 1.0f);
}

// ---------------- fusion head: linear + layernorm + scale + leaky into lf ----------------
// one block (256 threads) per graph row
__global__ void k_lin_ln(const float* __restrict__ in, int K, const float* __restrict__ W,
                         const float* __restrict__ bias, const float* __restrict__ alpha_p,
                         const float* __restrict__ beta_p, int mode,
                         float* __restrict__ lf, int off) {
  int g = blockIdx.x, j = threadIdx.x;
  const float* rowp = in + (size_t)g * K;
  float y = bias[j];
  for (int k = 0; k < K; ++k) y += rowp[k] * W[k * H + j];
  __shared__ float red[256];
  red[j] = y;
  __syncthreads();
  for (int s = 128; s > 0; s >>= 1) {
    if (j < s) red[j] += red[j + s];
    __syncthreads();
  }
  float mean = red[0] * (1.0f / 256.0f);
  __syncthreads();
  float d = y - mean;
  red[j] = d * d;
  __syncthreads();
  for (int s = 128; s > 0; s >>= 1) {
    if (j < s) red[j] += red[j + s];
    __syncthreads();
  }
  float var = red[0] * (1.0f / 256.0f);
  float a = sigmf(alpha_p[0]);
  float b = sigmf(beta_p[0]);
  float scale = (mode == 0) ? a : ((mode == 1) ? b : (1.0f - a - b));
  float v = scale * d * rsqrtf(var + EPS);
  lf[(size_t)g * (3 * H) + off + j] = leakyf(v);  // pre-apply leaky for fus1 input
}

__global__ void k_fus1(const float* __restrict__ lf, const float* __restrict__ W,
                       const float* __restrict__ bias, float* __restrict__ h1) {
  int t = blockIdx.x * blockDim.x + threadIdx.x;  // GG*256
  int g = t >> 8, j = t & 255;
  const float* rowp = lf + (size_t)g * (3 * H);
  float y = bias[j];
  for (int k = 0; k < 3 * H; ++k) y += rowp[k] * W[k * H + j];
  h1[t] = leakyf(y);
}

__global__ void k_fus2(const float* __restrict__ h1, const float* __restrict__ W,
                       const float* __restrict__ bias, float* __restrict__ out) {
  int g = blockIdx.x * blockDim.x + threadIdx.x;
  if (g >= GG) return;
  const float* rowp = h1 + (size_t)g * H;
  float y = bias[0];
  for (int k = 0; k < H; ++k) y += rowp[k] * W[k];
  out[g] = y;
}

// ---------------- launcher ----------------

static inline size_t alignup(size_t v) { return (v + 255) & ~(size_t)255; }

extern "C" void kernel_launch(void* const* d_in, const int* in_sizes, int n_in,
                              void* d_out, int out_size, void* d_ws, size_t ws_size,
                              hipStream_t stream) {
  const float* x      = (const float*)d_in[0];
  const int*   ei     = (const int*)d_in[1];
  const int*   batch  = (const int*)d_in[2];
  const float* gfeat  = (const float*)d_in[3];
  const float* smi    = (const float*)d_in[4];
  const float* Wl[3]  = {(const float*)d_in[5], (const float*)d_in[7], (const float*)d_in[9]};
  const float* bl[3]  = {(const float*)d_in[6], (const float*)d_in[8], (const float*)d_in[10]};
  const float* gnnW   = (const float*)d_in[11];
  const float* gnnb   = (const float*)d_in[12];
  const float* gfW    = (const float*)d_in[13];
  const float* gfb    = (const float*)d_in[14];
  const float* smiW   = (const float*)d_in[15];
  const float* smib   = (const float*)d_in[16];
  const float* f1W    = (const float*)d_in[17];
  const float* f1b    = (const float*)d_in[18];
  const float* f2W    = (const float*)d_in[19];
  const float* f2b    = (const float*)d_in[20];
  const float* alphap = (const float*)d_in[21];
  const float* betap  = (const float*)d_in[22];
  float* out = (float*)d_out;

  const int N = in_sizes[0] / H;       // 50000
  const int E = in_sizes[1] / 2;       // 800000
  const int NH = N * H;
  const int mtiles = N / 16;           // 3125
  const int waves = mtiles * 4;        // one wave per 16x64 output strip

  // workspace carve-out
  char* p = (char*)d_ws;
  size_t o = 0;
  bf16_t* Abf = (bf16_t*)(p + o); o = alignup(o + (size_t)NH * sizeof(bf16_t));
  float*  Hf  = (float*)(p + o);  o = alignup(o + (size_t)NH * sizeof(float));
  float*  agg = (float*)(p + o);  o = alignup(o + (size_t)NH * sizeof(float));
  float*  dis = (float*)(p + o);  o = alignup(o + (size_t)N * sizeof(float));
  bf16_t* Wp  = (bf16_t*)(p + o); o = alignup(o + (size_t)8 * 16 * 32 * 16 * sizeof(bf16_t));
  float*  ps  = (float*)(p + o);  o = alignup(o + (size_t)GG * H * sizeof(float));
  float*  cnt = (float*)(p + o);  o = alignup(o + (size_t)GG * sizeof(float));
  float*  lf  = (float*)(p + o);  o = alignup(o + (size_t)GG * 3 * H * sizeof(float));
  float*  h1  = (float*)(p + o);  o = alignup(o + (size_t)GG * H * sizeof(float));
  (void)ws_size; (void)n_in; (void)out_size;

  const int B = 256;
  const int* erow = ei;
  const int* ecol = ei + E;

  // degree -> dis = rsqrt(deg)
  k_fill1<<<(N + B - 1) / B, B, 0, stream>>>(dis, N);
  k_deg_edges<<<(E + B - 1) / B, B, 0, stream>>>(ecol, dis, E);
  k_dis<<<(N + B - 1) / B, B, 0, stream>>>(dis, N);

  // layer 1 input
  k_cvt<<<(NH + B - 1) / B, B, 0, stream>>>(x, Abf, NH);

  for (int layer = 0; layer < 3; ++layer) {
    k_pack_w<<<(8 * 16 * 32 * 16) / B, B, 0, stream>>>(Wl[layer], Wp);
    k_gemm_bf16<<<(waves * 32 + B - 1) / B, B, 0, stream>>>(Abf, Wp, Hf, mtiles);
    k_agg_init<<<(NH + B - 1) / B, B, 0, stream>>>(Hf, dis, bl[layer], agg, NH);
    k_agg_edges<<<((size_t)E * 32 + B - 1) / B, B, 0, stream>>>(erow, ecol, dis, Hf, agg, E);
    // leaky -> Hf (f32, pooling input for last layer) + Abf (bf16, next GEMM input)
    k_leaky_cvt<<<(NH + B - 1) / B, B, 0, stream>>>(agg, Hf, Abf, NH);
  }

  // global mean pool
  k_pool_init<<<(GG * H + B - 1) / B, B, 0, stream>>>(ps, cnt, GG * H);
  k_pool<<<(NH + B - 1) / B, B, 0, stream>>>(Hf, batch, ps, cnt, NH);
  k_pool_fin<<<GG * H / B, B, 0, stream>>>(ps, cnt);

  // three branch linears + layernorm + scale + leaky, concat order: [gnn, smi, graph]
  k_lin_ln<<<GG, 256, 0, stream>>>(ps,    H,  gnnW, gnnb, alphap, betap, 0, lf, 0);
  k_lin_ln<<<GG, 256, 0, stream>>>(smi,   SE, smiW, smib, alphap, betap, 1, lf, H);
  k_lin_ln<<<GG, 256, 0, stream>>>(gfeat, GF, gfW,  gfb,  alphap, betap, 2, lf, 2 * H);

  k_fus1<<<GG * H / B, B, 0, stream>>>(lf, f1W, f1b, h1);
  k_fus2<<<(GG + B - 1) / B, B, 0, stream>>>(h1, f2W, f2b, out);
}
